// Attention_18760417149505
// MI455X (gfx1250) — matrix operand order repaired and checked by reading, not compiled, not run on hardware
//
#include <hip/hip_runtime.h>

#define D_MODEL 768
#define SEQ     2048
#define BATCH   2
#define NHEAD   12
#define DK      64
#define TOK     (BATCH*SEQ)      // 4096 tokens
#define HEADSZ  (SEQ*DK)         // one head's flat [2048][64] slice
#define ATT_SCALE 0.125f         // 1/sqrt(64)

typedef __attribute__((ext_vector_type(8)))  __bf16 bf16x8;
typedef __attribute__((ext_vector_type(16))) __bf16 bf16x16;
typedef __attribute__((ext_vector_type(8)))  float  f32x8;
typedef unsigned int u32x4 __attribute__((ext_vector_type(4)));
typedef int          i32x4 __attribute__((ext_vector_type(4)));
typedef int          i32x8 __attribute__((ext_vector_type(8)));

#if defined(__has_builtin)
#if __has_builtin(__builtin_amdgcn_tensor_load_to_lds) && __has_builtin(__builtin_amdgcn_s_wait_tensorcnt)
#define USE_TDM 1
#endif
#endif
#ifndef USE_TDM
#define USE_TDM 0
#endif

// native RNE fp32 -> bf16 (lowers to v_cvt_pk_bf16_f32 on gfx1250)
__device__ __forceinline__ unsigned short bf_bits(float f) {
  __bf16 h = (__bf16)f;
  return __builtin_bit_cast(unsigned short, h);
}

// 16x bf16 fragment from two 16-byte LDS loads (ds_load_b128 each)
__device__ __forceinline__ bf16x16 frag16(const unsigned short* p0, const unsigned short* p1) {
  bf16x8 lo = *(const bf16x8*)p0;
  bf16x8 hi = *(const bf16x8*)p1;
  return __builtin_shufflevector(lo, hi, 0,1,2,3,4,5,6,7,8,9,10,11,12,13,14,15);
}

#if USE_TDM
// TDM: DMA one 2-D bf16 tile [tile_rows x 64] (128B rows) global->LDS with 16B
// LDS pad per row => LDS row stride 72 halves (144B). D# per CDNA5 ISA ch.8.
__device__ __forceinline__ void tdm_load_tile_bf16(unsigned int lds_byte_off,
                                                   const unsigned short* gsrc,
                                                   unsigned int tile_rows,
                                                   unsigned int row_stride_elems,
                                                   unsigned int tensor_rows) {
  unsigned long long ga = (unsigned long long)(uintptr_t)gsrc;
  u32x4 g0;
  g0[0] = 1u;                                            // count=1, user descriptor
  g0[1] = lds_byte_off;                                  // LDS dest (bytes)
  g0[2] = (unsigned int)(ga & 0xFFFFFFFFu);              // global_addr[31:0]
  g0[3] = (unsigned int)((ga >> 32) & 0x01FFFFFFu) | (2u << 30); // addr[56:32] | type=2
  i32x8 g1;
  // data_size=2B; pad on; pad_interval = 32 DW (code 4); pad_amount = 4 DW (code 3)
  g1[0] = (1 << 16) | (1 << 20) | (4 << 22) | (3 << 25);
  unsigned int td0 = row_stride_elems;                   // dim0 length (tile always in-range)
  unsigned int td1 = tensor_rows;
  g1[1] = (int)((td0 & 0xFFFFu) << 16);                  // tensor_dim0 lo (bits 79:48)
  g1[2] = (int)((td0 >> 16) | ((td1 & 0xFFFFu) << 16));  // dim0 hi | tensor_dim1 lo
  g1[3] = (int)((td1 >> 16) | (64u << 16));              // dim1 hi | tile_dim0 = 64
  g1[4] = (int)tile_rows;                                // tile_dim1 | tile_dim2=0
  g1[5] = (int)row_stride_elems;                         // tensor_dim0_stride lo
  g1[6] = 0;
  g1[7] = 0;
  i32x4 gz = {0, 0, 0, 0};
#if __clang_major__ >= 23
  i32x8 gz8 = {0, 0, 0, 0, 0, 0, 0, 0};
  __builtin_amdgcn_tensor_load_to_lds(g0, g1, gz, gz, gz8, 0);
#else
  __builtin_amdgcn_tensor_load_to_lds(g0, g1, gz, gz, 0);
#endif
}
#endif

// ---------------------------------------------------------------------------
// One-shot fp32 -> bf16 cast (x and the five weight matrices)
// ---------------------------------------------------------------------------
__global__ __launch_bounds__(256)
void cast_bf16_kernel(const float* __restrict__ in, unsigned short* __restrict__ out, int n) {
  int i = (blockIdx.x * 256 + threadIdx.x) * 4;
  if (i < n) {
    float4 f = *(const float4*)(in + i);
    ushort4 u;
    u.x = bf_bits(f.x); u.y = bf_bits(f.y); u.z = bf_bits(f.z); u.w = bf_bits(f.w);
    *(ushort4*)(out + i) = u;
  }
}

// shared WMMA body: one staged [128x64]x[128x64] K-block, 2 substeps of K=32
__device__ __forceinline__ void gemm_compute_block(const unsigned short* sAt,
                                                   const unsigned short* sBt,
                                                   int waveM, int waveN, int l16, int hl,
                                                   f32x8 acc[2][4]) {
  #pragma unroll
  for (int ks = 0; ks < 2; ++ks) {
    const int c0 = ks * 32;
    bf16x16 afrag[2], bfrag[4];
    #pragma unroll
    for (int mt = 0; mt < 2; ++mt) {
      // A 16x32: lanes<16 hold K 0..7 & 16..23; lanes>=16 hold K 8..15 & 24..31
      const unsigned short* p = sAt + (waveM*32 + mt*16 + l16)*72 + c0 + hl*8;
      afrag[mt] = frag16(p, p + 16);
    }
    #pragma unroll
    for (int nt = 0; nt < 4; ++nt) {
      // B 32x16: lanes<16 hold K 0..7 & 16..23; lanes>=16 hold K 8..15 & 24..31
      const unsigned short* p = sBt + (waveN*64 + nt*16 + l16)*72 + c0 + hl*8;
      bfrag[nt] = frag16(p, p + 16);
    }
    #pragma unroll
    for (int mt = 0; mt < 2; ++mt)
      #pragma unroll
      for (int nt = 0; nt < 4; ++nt)
        acc[mt][nt] = __builtin_amdgcn_wmma_f32_16x16x32_bf16(
            false, afrag[mt], false, bfrag[nt], (short)0, acc[mt][nt], false, false);
  }
}

// ---------------------------------------------------------------------------
// C[M,N] = A[M,K] @ W[N,K]^T + bias[N], bf16 operands, f32 accum.
// 256 thr (8 waves), 128x128 tile, staged K-block = 64.
// TDM path: double-buffered async pipeline (issue next block while computing).
// ---------------------------------------------------------------------------
template <bool OUT_BF16>
__global__ __launch_bounds__(256)
void gemm_bf16_kernel(const unsigned short* __restrict__ A, const unsigned short* __restrict__ W,
                      const float* __restrict__ bias, void* __restrict__ Cout,
                      int M, int N, int K)
{
  __shared__ __align__(16) unsigned short sA[2][128*72];  // [m][k] stride 72 halves
  __shared__ __align__(16) unsigned short sB[2][128*72];  // [n][k] = raw W rows

  const int tid  = threadIdx.x;
  const int lane = tid & 31, wave = tid >> 5;
  const int hl   = lane >> 4, l16 = lane & 15;
  const int waveM = wave >> 1, waveN = wave & 1;
  const int rowBase = blockIdx.y * 128;
  const int colBase = blockIdx.x * 128;

  f32x8 acc[2][4] = {};
  const int ksteps = K >> 6;                              // 64-wide K blocks

#if USE_TDM
  if (tid < 32) {                                         // TDM ignores EXEC; one wave issues
    tdm_load_tile_bf16((unsigned int)(uintptr_t)&sA[0][0],
                       A + (size_t)rowBase * K, 128, K, M - rowBase);
    tdm_load_tile_bf16((unsigned int)(uintptr_t)&sB[0][0],
                       W + (size_t)colBase * K, 128, K, N - colBase);
    __builtin_amdgcn_s_wait_tensorcnt(0);
  }
  __syncthreads();                                        // block 0 visible

  for (int kt = 0; kt < ksteps; ++kt) {
    const int cur = kt & 1;
    if (tid < 32 && (kt + 1) < ksteps) {                  // prefetch next block async
      const int k0 = (kt + 1) << 6;
      tdm_load_tile_bf16((unsigned int)(uintptr_t)&sA[cur ^ 1][0],
                         A + (size_t)rowBase * K + k0, 128, K, M - rowBase);
      tdm_load_tile_bf16((unsigned int)(uintptr_t)&sB[cur ^ 1][0],
                         W + (size_t)colBase * K + k0, 128, K, N - colBase);
    }
    gemm_compute_block(&sA[cur][0], &sB[cur][0], waveM, waveN, l16, hl, acc);
    __syncthreads();                                      // all done reading buf[cur]
    if (tid < 32) __builtin_amdgcn_s_wait_tensorcnt(0);   // next block landed
    __syncthreads();                                      // publish to all waves
  }
#else
  for (int kt = 0; kt < ksteps; ++kt) {
    const int k0 = kt << 6;
    __syncthreads();
    #pragma unroll
    for (int i = 0; i < 4; ++i) {                         // 1024 b128 chunks / 256 thr
      int e = i*256 + tid;
      int r = e >> 3, c = (e & 7) * 8;
      *(uint4*)&sA[0][r*72 + c] = *(const uint4*)&A[(size_t)(rowBase + r)*K + k0 + c];
      *(uint4*)&sB[0][r*72 + c] = *(const uint4*)&W[(size_t)(colBase + r)*K + k0 + c];
    }
    __syncthreads();
    gemm_compute_block(&sA[0][0], &sB[0][0], waveM, waveN, l16, hl, acc);
  }
#endif

  // C/D layout: col = lane%16, row = vgpr + 8*(lane>=16)
  #pragma unroll
  for (int mt = 0; mt < 2; ++mt) {
    #pragma unroll
    for (int nt = 0; nt < 4; ++nt) {
      int col  = colBase + waveN*64 + nt*16 + l16;
      int row0 = rowBase + waveM*32 + mt*16 + hl*8;
      float bv = bias[col];
      #pragma unroll
      for (int r = 0; r < 8; ++r) {
        float v = acc[mt][nt][r] + bv;
        if (OUT_BF16) ((unsigned short*)Cout)[(size_t)(row0 + r)*N + col] = bf_bits(v);
        else          ((float*)Cout)[(size_t)(row0 + r)*N + col] = v;
      }
    }
  }
}

// ---------------------------------------------------------------------------
// Per (b,h): Mt[d][e] = SCALE * (K_h^T @ V_h)^T  (64x64 bf16, [d][e] row-major)
// 64-seq chunks transposed into LDS; 8 waves own 2 output tiles each.
// ---------------------------------------------------------------------------
__global__ __launch_bounds__(256)
void kv_reduce_kernel(const unsigned short* __restrict__ kbuf,
                      const unsigned short* __restrict__ vbuf,
                      unsigned short* __restrict__ mbuf)
{
  __shared__ __align__(16) unsigned short sKT[64*72];   // [d][s] (s chunk of 64)
  __shared__ __align__(16) unsigned short sVT[64*72];

  const int bh = blockIdx.x;
  const unsigned short* Kh = kbuf + (size_t)bh * HEADSZ;
  const unsigned short* Vh = vbuf + (size_t)bh * HEADSZ;

  const int tid  = threadIdx.x;
  const int lane = tid & 31, wave = tid >> 5;
  const int hl   = lane >> 4, l16 = lane & 15;
  const int mt     = wave >> 1;                         // e-tile (rows of M)
  const int ntBase = (wave & 1) * 2;                    // d-tile base

  f32x8 acc[2] = {};

  for (int it = 0; it < SEQ/64; ++it) {
    const int s0 = it * 64;
    __syncthreads();
    #pragma unroll
    for (int i = 0; i < 2; ++i) {                       // 64x64 chunk: 2 b128/thr/matrix
      int e = i*256 + tid;
      int s = e >> 3, d0 = (e & 7) * 8;
      union { uint4 v; unsigned short h[8]; } kq, vq;
      kq.v = *(const uint4*)&Kh[(size_t)(s0 + s)*DK + d0];
      vq.v = *(const uint4*)&Vh[(size_t)(s0 + s)*DK + d0];
      #pragma unroll
      for (int j = 0; j < 8; ++j) {
        sKT[(d0 + j)*72 + s] = kq.h[j];
        sVT[(d0 + j)*72 + s] = vq.h[j];
      }
    }
    __syncthreads();

    #pragma unroll
    for (int ks = 0; ks < 2; ++ks) {
      const int c0 = ks * 32;
      const unsigned short* pa = sKT + (mt*16 + l16)*72 + c0 + hl*8;
      bf16x16 afrag = frag16(pa, pa + 16);
      #pragma unroll
      for (int j = 0; j < 2; ++j) {
        const unsigned short* pb = sVT + ((ntBase + j)*16 + l16)*72 + c0 + hl*8;
        bf16x16 bfrag = frag16(pb, pb + 16);
        acc[j] = __builtin_amdgcn_wmma_f32_16x16x32_bf16(
            false, afrag, false, bfrag, (short)0, acc[j], false, false);
      }
    }
  }

  #pragma unroll
  for (int j = 0; j < 2; ++j) {
    int d  = (ntBase + j)*16 + l16;
    int e0 = mt*16 + hl*8;
    #pragma unroll
    for (int r = 0; r < 8; ++r)
      mbuf[(size_t)bh*4096 + d*64 + (e0 + r)] = bf_bits(acc[j][r] * ATT_SCALE);
  }
}

// ---------------------------------------------------------------------------
// Per (b,h, 128-row tile): P = Q_tile @ M (K=64), softmax over dk=64, bf16 out.
// Q and M tiles staged through the TDM (same 128B-row + 16B-pad layout as GEMM).
// ---------------------------------------------------------------------------
__global__ __launch_bounds__(256)
void qm_softmax_kernel(const unsigned short* __restrict__ qbuf,
                       const unsigned short* __restrict__ mbuf,
                       unsigned short* __restrict__ attn)
{
  __shared__ __align__(16) unsigned short sQ[128*72];   // [s][e]
  __shared__ __align__(16) unsigned short sM[64*72];    // [d][e]
  __shared__ float sP[128*64];

  const int bh    = blockIdx.y;
  const int sBase = blockIdx.x * 128;
  const unsigned short* Qh = qbuf + (size_t)bh * HEADSZ;

  const int tid  = threadIdx.x;
  const int lane = tid & 31, wave = tid >> 5;
  const int hl   = lane >> 4, l16 = lane & 15;

#if USE_TDM
  if (tid < 32) {
    tdm_load_tile_bf16((unsigned int)(uintptr_t)&sQ[0],
                       Qh + (size_t)sBase * DK, 128, DK, SEQ - sBase);
    tdm_load_tile_bf16((unsigned int)(uintptr_t)&sM[0],
                       mbuf + (size_t)bh * 4096, 64, DK, 64);
    __builtin_amdgcn_s_wait_tensorcnt(0);
  }
#else
  #pragma unroll
  for (int i = 0; i < 4; ++i) {                         // 128x64 Q: 1024 b128 chunks
    int e = i*256 + tid;
    int r = e >> 3, c = (e & 7) * 8;
    *(uint4*)&sQ[r*72 + c] = *(const uint4*)&Qh[(size_t)(sBase + r)*DK + c];
  }
  #pragma unroll
  for (int i = 0; i < 2; ++i) {                         // 64x64 M^T: 512 b128 chunks
    int e = i*256 + tid;
    int r = e >> 3, c = (e & 7) * 8;
    *(uint4*)&sM[r*72 + c] = *(const uint4*)&mbuf[(size_t)bh*4096 + r*64 + c];
  }
#endif
  __syncthreads();

  f32x8 acc[4] = {};
  #pragma unroll
  for (int ks = 0; ks < 2; ++ks) {
    const int k0 = ks*32;
    const unsigned short* pa = sQ + (wave*16 + l16)*72 + k0 + hl*8;
    bf16x16 afrag = frag16(pa, pa + 16);
    #pragma unroll
    for (int nt = 0; nt < 4; ++nt) {
      const unsigned short* pb = sM + (nt*16 + l16)*72 + k0 + hl*8;
      bf16x16 bfrag = frag16(pb, pb + 16);
      acc[nt] = __builtin_amdgcn_wmma_f32_16x16x32_bf16(
          false, afrag, false, bfrag, (short)0, acc[nt], false, false);
    }
  }

  #pragma unroll
  for (int nt = 0; nt < 4; ++nt) {
    int row0 = wave*16 + hl*8;
    int col  = nt*16 + l16;
    #pragma unroll
    for (int r = 0; r < 8; ++r)
      sP[(row0 + r)*64 + col] = acc[nt][r];
  }
  __syncthreads();

  if (tid < 128) {                                      // softmax over dk=64
    const float* pr = sP + tid*64;
    float m = pr[0];
    #pragma unroll
    for (int d = 1; d < 64; ++d) m = fmaxf(m, pr[d]);
    float sum = 0.f;
    #pragma unroll
    for (int d = 0; d < 64; ++d) sum += __expf(pr[d] - m);
    float inv = 1.f / sum;
    unsigned short* outp = attn + (size_t)bh*HEADSZ + (size_t)(sBase + tid)*DK;
    #pragma unroll
    for (int g = 0; g < 8; ++g) {                       // pack 8 bf16 -> one b128 store
      unsigned int w[4];
      #pragma unroll
      for (int p = 0; p < 4; ++p) {
        unsigned short a = bf_bits(__expf(pr[g*8 + 2*p]     - m) * inv);
        unsigned short b = bf_bits(__expf(pr[g*8 + 2*p + 1] - m) * inv);
        w[p] = (unsigned int)a | ((unsigned int)b << 16);
      }
      uint4 o = {w[0], w[1], w[2], w[3]};
      *(uint4*)(outp + g*8) = o;
    }
  }
}

// ---------------------------------------------------------------------------
extern "C" void kernel_launch(void* const* d_in, const int* in_sizes, int n_in,
                              void* d_out, int out_size, void* d_ws, size_t ws_size,
                              hipStream_t stream)
{
  const float* x   = (const float*)d_in[0];
  const float* Wq  = (const float*)d_in[1];
  const float* bq  = (const float*)d_in[2];
  const float* Wk  = (const float*)d_in[3];
  const float* bk  = (const float*)d_in[4];
  const float* Wv  = (const float*)d_in[5];
  const float* bv  = (const float*)d_in[6];
  const float* Wo  = (const float*)d_in[7];
  const float* bo  = (const float*)d_in[8];
  const float* Wf  = (const float*)d_in[9];
  const float* bf_ = (const float*)d_in[10];
  float* out = (float*)d_out;

  const size_t PLANE = (size_t)TOK * D_MODEL;           // 3,145,728
  const size_t WSZ   = (size_t)D_MODEL * D_MODEL;       //   589,824
  unsigned short* xb  = (unsigned short*)d_ws;
  unsigned short* wqb = xb  + PLANE;
  unsigned short* wkb = wqb + WSZ;
  unsigned short* wvb = wkb + WSZ;
  unsigned short* wob = wvb + WSZ;
  unsigned short* wfb = wob + WSZ;
  unsigned short* qb  = wfb + WSZ;
  unsigned short* kb  = qb  + PLANE;
  unsigned short* vb  = kb  + PLANE;
  unsigned short* ab  = vb  + PLANE;
  unsigned short* t1  = ab  + PLANE;
  unsigned short* mb  = t1  + PLANE;                    // 24 * 64*64 bf16

  cast_bf16_kernel<<<(int)(PLANE/1024), 256, 0, stream>>>(x,  xb,  (int)PLANE);
  cast_bf16_kernel<<<(int)(WSZ/1024),   256, 0, stream>>>(Wq, wqb, (int)WSZ);
  cast_bf16_kernel<<<(int)(WSZ/1024),   256, 0, stream>>>(Wk, wkb, (int)WSZ);
  cast_bf16_kernel<<<(int)(WSZ/1024),   256, 0, stream>>>(Wv, wvb, (int)WSZ);
  cast_bf16_kernel<<<(int)(WSZ/1024),   256, 0, stream>>>(Wo, wob, (int)WSZ);
  cast_bf16_kernel<<<(int)(WSZ/1024),   256, 0, stream>>>(Wf, wfb, (int)WSZ);

  dim3 gg(D_MODEL/128, TOK/128);                        // (6, 32)
  gemm_bf16_kernel<true ><<<gg, 256, 0, stream>>>(xb, wqb, bq, qb, TOK, D_MODEL, D_MODEL);
  gemm_bf16_kernel<true ><<<gg, 256, 0, stream>>>(xb, wkb, bk, kb, TOK, D_MODEL, D_MODEL);
  gemm_bf16_kernel<true ><<<gg, 256, 0, stream>>>(xb, wvb, bv, vb, TOK, D_MODEL, D_MODEL);

  kv_reduce_kernel<<<BATCH*NHEAD, 256, 0, stream>>>(kb, vb, mb);

  dim3 gq(SEQ/128, BATCH*NHEAD);                        // (16, 24)
  qm_softmax_kernel<<<gq, 256, 0, stream>>>(qb, mb, ab);

  gemm_bf16_kernel<true ><<<gg, 256, 0, stream>>>(ab, wob, bo, t1, TOK, D_MODEL, D_MODEL);
  gemm_bf16_kernel<false><<<gg, 256, 0, stream>>>(t1, wfb, bf_, out, TOK, D_MODEL, D_MODEL);
}
